// MultiHeadLatentAttention_79259326480572
// MI455X (gfx1250) — compile-verified, matrix operand order checked by reading, not run on hardware
//
#include <hip/hip_runtime.h>

// ---------------------------------------------------------------------------
// Types for CDNA5 WMMA (gfx1250, wave32)
// ---------------------------------------------------------------------------
typedef __attribute__((ext_vector_type(16))) __bf16 v16bf;
typedef __attribute__((ext_vector_type(8)))  float  v8f;
typedef __attribute__((ext_vector_type(4))) unsigned tdm_u32x4;
typedef __attribute__((ext_vector_type(8))) int      tdm_i32x8;
typedef __attribute__((ext_vector_type(4))) int      tdm_i32x4;

#if defined(__has_builtin)
#if __has_builtin(__builtin_amdgcn_tensor_load_to_lds) && \
    __has_builtin(__builtin_amdgcn_s_wait_tensorcnt)
#define HAS_TDM 1
#endif
#endif
#ifndef HAS_TDM
#define HAS_TDM 0
#endif

union Frag {
  v16bf v;
  uint4 q[2];
};

__device__ __forceinline__ unsigned short f2bf(float f) {
  unsigned u = __float_as_uint(f);
  unsigned r = (u + 0x7FFFu + ((u >> 16) & 1u)) >> 16;  // round-to-nearest-even
  return (unsigned short)r;
}

__device__ __forceinline__ unsigned pack2(float a, float b) {
  return (unsigned)f2bf(a) | ((unsigned)f2bf(b) << 16);
}

__device__ __forceinline__ v8f v8f_zero() {
  v8f z;
#pragma unroll
  for (int i = 0; i < 8; ++i) z[i] = 0.0f;
  return z;
}

#if HAS_TDM
// Issue a TDM 2D tile load: bf16 tile (tile_w x tile_h), row stride
// stride_elems, from global 'gaddr' into LDS byte offset 'lds_off'.
// Descriptor packing per CDNA5 ISA 8.3/8.4 (D# group0/group1).
// This toolchain's builtin takes 6 args:
//   (uint32x4 g0, int32x8 g1, int32x4 g2, int32x4 g3, int32x8 pad, i32 cpol)
__device__ __forceinline__ void tdm_load_2d(const void* gaddr, unsigned lds_off,
                                            unsigned tile_w, unsigned tile_h,
                                            unsigned stride_elems) {
  unsigned long long ga = (unsigned long long)gaddr;
  tdm_u32x4 g0;
  g0[0] = 1u;                                   // count=1 (valid descriptor)
  g0[1] = lds_off;                              // lds_addr (bytes)
  g0[2] = (unsigned)(ga & 0xFFFFFFFFu);         // global_addr[31:0]
  g0[3] = (unsigned)((ga >> 32) & 0x1FFFFFFu)   // global_addr[56:32]
        | (2u << 30);                           // type=2 ("image")
  tdm_i32x8 g1;
  g1[0] = (int)(1u << 16);                      // data_size=1 -> 2 bytes
  g1[1] = (int)((tile_w & 0xFFFFu) << 16);      // tensor_dim0[15:0]
  g1[2] = (int)((tile_w >> 16) & 0xFFFFu)       // tensor_dim0[31:16]
        | (int)((tile_h & 0xFFFFu) << 16);      // tensor_dim1[15:0]
  g1[3] = (int)((tile_h >> 16) & 0xFFFFu)       // tensor_dim1[31:16]
        | (int)((tile_w & 0xFFFFu) << 16);      // tile_dim0
  g1[4] = (int)(tile_h & 0xFFFFu);              // tile_dim1 (tile_dim2=0)
  g1[5] = (int)stride_elems;                    // tensor_dim0_stride[31:0]
  g1[6] = 0;                                    // stride hi / dim1_stride lo
  g1[7] = 0;
  tdm_i32x4 z4 = {0, 0, 0, 0};
  tdm_i32x8 z8 = {0, 0, 0, 0, 0, 0, 0, 0};
  __builtin_amdgcn_tensor_load_to_lds(g0, g1, z4, z4, z8, 0);
}
#endif

// ---------------------------------------------------------------------------
// fp32 -> bf16 elementwise convert (for x)
// ---------------------------------------------------------------------------
__global__ __launch_bounds__(256) void cvt_f32_bf16(const float* __restrict__ in,
                                                    unsigned short* __restrict__ out,
                                                    int n) {
  int i = blockIdx.x * 256 + threadIdx.x;
  if (i < n) out[i] = f2bf(in[i]);
}

// ---------------------------------------------------------------------------
// fp32 [R][C] -> bf16 transposed [C][R]  (weights become N-major "Bt")
// ---------------------------------------------------------------------------
__global__ __launch_bounds__(256) void transpose_f32_bf16(const float* __restrict__ in,
                                                          unsigned short* __restrict__ out,
                                                          int R, int C) {
  __shared__ float tile[32][33];
  int tx = threadIdx.x & 31;
  int ty = threadIdx.x >> 5;  // 0..7
  int r0 = blockIdx.y * 32, c0 = blockIdx.x * 32;
#pragma unroll
  for (int i = 0; i < 32; i += 8) {
    int r = r0 + ty + i, c = c0 + tx;
    tile[ty + i][tx] = (r < R && c < C) ? in[(size_t)r * C + c] : 0.0f;
  }
  __syncthreads();
#pragma unroll
  for (int i = 0; i < 32; i += 8) {
    int orow = c0 + ty + i;  // out row = input col
    int ocol = r0 + tx;      // out col = input row
    if (orow < C && ocol < R) out[(size_t)orow * R + ocol] = f2bf(tile[tx][ty + i]);
  }
}

// ---------------------------------------------------------------------------
// bf16 WMMA GEMM:  C = epilogue( A[MxK] @ Bt[NxK]^T )
// Block = 4 waves; each wave computes 32 rows x 64 cols (8 WMMA / K-step).
// Block tile = 128 x 64.  B tile (64 rows x 32 K) staged in LDS,
// double-buffered via the Tensor Data Mover (wave 0 issues, s_wait_tensorcnt
// + workgroup barrier for handoff).  flags: 1=ReLU, 2=bf16 transposed store,
// 4=f32 store (+bias).  Default: bf16 row-major store.
// ---------------------------------------------------------------------------
#define GF_RELU   1
#define GF_TSTORE 2
#define GF_F32OUT 4

__device__ __forceinline__ void stage_b_tile(const unsigned short* Bt, int col0,
                                             int ldb, int kb,
                                             unsigned short (*buf)[32],
                                             int tid, int wave) {
#if HAS_TDM
  (void)tid;
  if (wave == 0) {
    tdm_load_2d(Bt + (size_t)col0 * ldb + kb, (unsigned)(size_t)&buf[0][0],
                /*tile_w=*/32, /*tile_h=*/64, (unsigned)ldb);
  }
#else
  (void)wave;
  // cooperative copy: 256 chunks of 16B, 128 threads x 2
#pragma unroll
  for (int p = 0; p < 2; ++p) {
    int chunk = tid + p * 128;
    int r = chunk >> 2, c = (chunk & 3) * 8;
    *(uint4*)&buf[r][c] = *(const uint4*)(Bt + (size_t)(col0 + r) * ldb + kb + c);
  }
#endif
}

__device__ __forceinline__ void stage_wait(int wave) {
#if HAS_TDM
  if (wave == 0) __builtin_amdgcn_s_wait_tensorcnt((short)0);
#else
  (void)wave;
#endif
  __syncthreads();
}

__global__ __launch_bounds__(128) void gemm_bf16_wmma(
    const unsigned short* __restrict__ A, const unsigned short* __restrict__ Bt,
    void* __restrict__ Cout, const float* __restrict__ bias,
    int K, int lda, int ldb, int ldc, float scale, int flags) {
  __shared__ unsigned short lds_b[2][64][32];  // double-buffered B tile (8KB)

  const int tid   = threadIdx.x;
  const int lane  = tid & 31;
  const int wave  = tid >> 5;
  const int l16   = lane & 15;
  const int half8 = (lane < 16) ? 0 : 8;

  const int row0 = blockIdx.y * 128 + wave * 32;
  const int col0 = blockIdx.x * 64;

  v8f acc[2][4];
#pragma unroll
  for (int rt = 0; rt < 2; ++rt)
#pragma unroll
    for (int t = 0; t < 4; ++t) acc[rt][t] = v8f_zero();

  const unsigned short* arow0 = A + (size_t)(row0 + l16) * lda + half8;
  const unsigned short* arow1 = arow0 + (size_t)16 * lda;

  // prologue: stage first B tile
  stage_b_tile(Bt, col0, ldb, 0, lds_b[0], tid, wave);
  stage_wait(wave);

  int cur = 0;
  for (int kb = 0; kb < K; kb += 32) {
    const bool more = (kb + 32) < K;
    if (more) stage_b_tile(Bt, col0, ldb, kb + 32, lds_b[cur ^ 1], tid, wave);

    Frag af0, af1;
    af0.q[0] = *(const uint4*)(arow0 + kb);
    af0.q[1] = *(const uint4*)(arow0 + kb + 16);
    af1.q[0] = *(const uint4*)(arow1 + kb);
    af1.q[1] = *(const uint4*)(arow1 + kb + 16);
    __builtin_prefetch(arow0 + kb + 64, 0, 3);
    __builtin_prefetch(arow1 + kb + 64, 0, 3);

    // load all four B fragments into distinct register octets first, so the
    // eight ds_load_b128 can issue as one clause and the eight WMMAs run
    // back-to-back behind a single dscnt wait
    Frag bfr[4];
#pragma unroll
    for (int t = 0; t < 4; ++t) {
      const unsigned short* bl = &lds_b[cur][t * 16 + l16][half8];
      bfr[t].q[0] = *(const uint4*)(bl);
      bfr[t].q[1] = *(const uint4*)(bl + 16);
    }
#pragma unroll
    for (int t = 0; t < 4; ++t) {
      acc[0][t] = __builtin_amdgcn_wmma_f32_16x16x32_bf16(
          false, af0.v, false, bfr[t].v, (short)0, acc[0][t], false, false);
      acc[1][t] = __builtin_amdgcn_wmma_f32_16x16x32_bf16(
          false, af1.v, false, bfr[t].v, (short)0, acc[1][t], false, false);
    }

    if (more) {
      stage_wait(wave);  // next buffer ready for everyone
      cur ^= 1;
    }
  }

  // ---- epilogue ----
  const bool relu = flags & GF_RELU;
#pragma unroll
  for (int rt = 0; rt < 2; ++rt) {
    const int rbase = row0 + rt * 16;
#pragma unroll
    for (int t = 0; t < 4; ++t) {
      int n = col0 + t * 16 + l16;
      float vals[8];
#pragma unroll
      for (int r = 0; r < 8; ++r) {
        float v = acc[rt][t][r] * scale;
        if (relu) v = fmaxf(v, 0.0f);
        vals[r] = v;
      }
      if (flags & GF_F32OUT) {
        float* C = (float*)Cout;
        float bv = bias ? bias[n] : 0.0f;
#pragma unroll
        for (int r = 0; r < 8; ++r) {
          int m = rbase + r + half8;
          C[(size_t)m * ldc + n] = vals[r] + bv;
        }
      } else if (flags & GF_TSTORE) {
        // lane's 8 values are consecutive rows at fixed column n
        unsigned short* Ct = (unsigned short*)Cout;
        uint4 pk;
        pk.x = pack2(vals[0], vals[1]);
        pk.y = pack2(vals[2], vals[3]);
        pk.z = pack2(vals[4], vals[5]);
        pk.w = pack2(vals[6], vals[7]);
        *(uint4*)(Ct + (size_t)n * ldc + rbase + half8) = pk;
      } else {
        unsigned short* C = (unsigned short*)Cout;
#pragma unroll
        for (int r = 0; r < 8; ++r) {
          int m = rbase + r + half8;
          C[(size_t)m * ldc + n] = f2bf(vals[r]);
        }
      }
    }
  }
}

// ---------------------------------------------------------------------------
// Flash attention (bf16 WMMA).  q: [b*2048][1024] (scale prefolded),
// k: [b*2048][1024], vT: [1024][4096] ( [h*64+d][b*2048+n] ),
// out ab: [b*2048][1024] bf16.
// Block = 4 waves x 16 q-rows; 64-key chunks -> 16 WMMA per chunk.
// ---------------------------------------------------------------------------
__global__ __launch_bounds__(128) void mla_attention(
    const unsigned short* __restrict__ qb, const unsigned short* __restrict__ kb,
    const unsigned short* __restrict__ vT, unsigned short* __restrict__ ab) {
  const int NSEQ = 2048, DIM = 1024;
  __shared__ unsigned short lds_p[4][16][64];  // per-wave P tile (16x64 bf16)

  const int lane  = threadIdx.x & 31;
  const int wave  = threadIdx.x >> 5;
  const int l16   = lane & 15;
  const int half8 = (lane < 16) ? 0 : 8;

  const int bh = blockIdx.x;
  const int b  = bh >> 4;
  const int h  = bh & 15;
  const int qr0 = blockIdx.y * 64 + wave * 16;

  // q fragments for this wave's 16 rows (d = 64 -> two K=32 fragments)
  Frag qf[2];
  const unsigned short* qrow =
      qb + (size_t)(b * NSEQ + qr0 + l16) * DIM + h * 64 + half8;
  qf[0].q[0] = *(const uint4*)(qrow);
  qf[0].q[1] = *(const uint4*)(qrow + 16);
  qf[1].q[0] = *(const uint4*)(qrow + 32);
  qf[1].q[1] = *(const uint4*)(qrow + 48);

  v8f o[4];
#pragma unroll
  for (int t = 0; t < 4; ++t) o[t] = v8f_zero();
  float mrun[8], lsum[8];
#pragma unroll
  for (int r = 0; r < 8; ++r) { mrun[r] = -3.0e38f; lsum[r] = 0.0f; }

  for (int kc = 0; kc < NSEQ; kc += 64) {
    // ---- S = q @ k^T for 64 keys (four 16x16 tiles) ----
    v8f s[4];
#pragma unroll
    for (int t = 0; t < 4; ++t) {
      const unsigned short* krow =
          kb + (size_t)(b * NSEQ + kc + t * 16 + l16) * DIM + h * 64 + half8;
      Frag k0, k1;
      k0.q[0] = *(const uint4*)(krow);
      k0.q[1] = *(const uint4*)(krow + 16);
      k1.q[0] = *(const uint4*)(krow + 32);
      k1.q[1] = *(const uint4*)(krow + 48);
      v8f c = v8f_zero();
      c = __builtin_amdgcn_wmma_f32_16x16x32_bf16(false, qf[0].v, false, k0.v,
                                                  (short)0, c, false, false);
      c = __builtin_amdgcn_wmma_f32_16x16x32_bf16(false, qf[1].v, false, k1.v,
                                                  (short)0, c, false, false);
      s[t] = c;
    }

    // ---- online softmax: row stats live on element r across 16 lanes ----
    float nm[8], al[8];
#pragma unroll
    for (int r = 0; r < 8; ++r) {
      float mx = fmaxf(fmaxf(s[0][r], s[1][r]), fmaxf(s[2][r], s[3][r]));
      mx = fmaxf(mx, __shfl_xor(mx, 1));
      mx = fmaxf(mx, __shfl_xor(mx, 2));
      mx = fmaxf(mx, __shfl_xor(mx, 4));
      mx = fmaxf(mx, __shfl_xor(mx, 8));
      float newm = fmaxf(mrun[r], mx);
      al[r] = __expf(mrun[r] - newm);
      nm[r] = newm;
      mrun[r] = newm;
    }
#pragma unroll
    for (int r = 0; r < 8; ++r) {
      float rs = 0.0f;
#pragma unroll
      for (int t = 0; t < 4; ++t) {
        float p = __expf(s[t][r] - nm[r]);
        s[t][r] = p;
        rs += p;
      }
      rs += __shfl_xor(rs, 1);
      rs += __shfl_xor(rs, 2);
      rs += __shfl_xor(rs, 4);
      rs += __shfl_xor(rs, 8);
      lsum[r] = lsum[r] * al[r] + rs;
    }
#pragma unroll
    for (int t = 0; t < 4; ++t)
#pragma unroll
      for (int r = 0; r < 8; ++r) o[t][r] *= al[r];

    // ---- reshape P (C-layout) -> A-fragments via per-wave LDS ----
#pragma unroll
    for (int t = 0; t < 4; ++t)
#pragma unroll
      for (int r = 0; r < 8; ++r)
        lds_p[wave][r + half8][t * 16 + l16] = f2bf(s[t][r]);
    asm volatile("s_wait_dscnt 0" ::: "memory");  // LDS in-order per wave
    Frag pf[2];
    const unsigned short* lp = &lds_p[wave][l16][half8];
    pf[0].q[0] = *(const uint4*)(lp);
    pf[0].q[1] = *(const uint4*)(lp + 16);
    pf[1].q[0] = *(const uint4*)(lp + 32);
    pf[1].q[1] = *(const uint4*)(lp + 48);

    // ---- O += P @ V  (Bt fragments from pre-transposed vT) ----
#pragma unroll
    for (int dt = 0; dt < 4; ++dt) {
      const unsigned short* vrow =
          vT + (size_t)(h * 64 + dt * 16 + l16) * 4096 + b * NSEQ + kc + half8;
      Frag v0, v1;
      v0.q[0] = *(const uint4*)(vrow);
      v0.q[1] = *(const uint4*)(vrow + 16);
      v1.q[0] = *(const uint4*)(vrow + 32);
      v1.q[1] = *(const uint4*)(vrow + 48);
      o[dt] = __builtin_amdgcn_wmma_f32_16x16x32_bf16(false, pf[0].v, false, v0.v,
                                                      (short)0, o[dt], false, false);
      o[dt] = __builtin_amdgcn_wmma_f32_16x16x32_bf16(false, pf[1].v, false, v1.v,
                                                      (short)0, o[dt], false, false);
    }
  }

  // ---- finalize: divide by lsum, store bf16 row-major [b*2048+m][h*64+d] ----
#pragma unroll
  for (int dt = 0; dt < 4; ++dt) {
    int col = h * 64 + dt * 16 + l16;
#pragma unroll
    for (int r = 0; r < 8; ++r) {
      int m = b * NSEQ + qr0 + r + half8;
      ab[(size_t)m * DIM + col] = f2bf(o[dt][r] / lsum[r]);
    }
  }
}

// ---------------------------------------------------------------------------
// Host orchestration
// ---------------------------------------------------------------------------
extern "C" void kernel_launch(void* const* d_in, const int* in_sizes, int n_in,
                              void* d_out, int out_size, void* d_ws, size_t ws_size,
                              hipStream_t stream) {
  (void)in_sizes; (void)n_in; (void)out_size; (void)ws_size;
  const float* x   = (const float*)d_in[0];
  const float* Wq  = (const float*)d_in[1];
  const float* Wkv = (const float*)d_in[2];
  const float* Wk1 = (const float*)d_in[3];
  const float* Wk2 = (const float*)d_in[4];
  const float* Wv1 = (const float*)d_in[5];
  const float* Wv2 = (const float*)d_in[6];
  const float* Wo  = (const float*)d_in[7];
  const float* bo  = (const float*)d_in[8];
  float* out = (float*)d_out;

  const int M = 4096;  // b*n = 2*2048
  char* ws = (char*)d_ws;
  size_t off = 0;
  auto alloc = [&](size_t elems) {
    void* p = ws + off;
    off += ((elems * 2) + 255) & ~(size_t)255;  // bf16 elements
    return (unsigned short*)p;
  };

  unsigned short* xb    = alloc((size_t)M * 1024);
  unsigned short* WqT   = alloc(1024 * 1024);
  unsigned short* WkvT  = alloc(2048 * 1024);
  unsigned short* Wk1T  = alloc(512 * 1024);
  unsigned short* Wk2T  = alloc(1024 * 512);
  unsigned short* Wv1T  = alloc(512 * 1024);
  unsigned short* Wv2T  = alloc(1024 * 512);
  unsigned short* WoT   = alloc(1024 * 1024);
  unsigned short* qbuf  = alloc((size_t)M * 1024);
  unsigned short* kvb   = alloc((size_t)M * 2048);
  unsigned short* hk    = alloc((size_t)M * 512);
  unsigned short* hv    = alloc((size_t)M * 512);
  unsigned short* kbuf  = alloc((size_t)M * 1024);
  unsigned short* vTb   = alloc((size_t)1024 * M);
  unsigned short* ab    = alloc((size_t)M * 1024);

  // --- precision conversion + weight transposes (bf16, N-major) ---
  cvt_f32_bf16<<<(M * 1024 + 255) / 256, 256, 0, stream>>>(x, xb, M * 1024);
  transpose_f32_bf16<<<dim3(1024 / 32, 1024 / 32), 256, 0, stream>>>(Wq,  WqT,  1024, 1024);
  transpose_f32_bf16<<<dim3(2048 / 32, 1024 / 32), 256, 0, stream>>>(Wkv, WkvT, 1024, 2048);
  transpose_f32_bf16<<<dim3(512 / 32, 1024 / 32), 256, 0, stream>>>(Wk1, Wk1T, 1024, 512);
  transpose_f32_bf16<<<dim3(1024 / 32, 512 / 32), 256, 0, stream>>>(Wk2, Wk2T, 512, 1024);
  transpose_f32_bf16<<<dim3(512 / 32, 1024 / 32), 256, 0, stream>>>(Wv1, Wv1T, 1024, 512);
  transpose_f32_bf16<<<dim3(1024 / 32, 512 / 32), 256, 0, stream>>>(Wv2, Wv2T, 512, 1024);
  transpose_f32_bf16<<<dim3(1024 / 32, 1024 / 32), 256, 0, stream>>>(Wo,  WoT,  1024, 1024);

  const dim3 blk(128);
  // q = (x @ Wq) * 1/sqrt(64)   (scale folded in)
  gemm_bf16_wmma<<<dim3(1024 / 64, M / 128), blk, 0, stream>>>(
      xb, WqT, qbuf, nullptr, 1024, 1024, 1024, 1024, 0.125f, 0);
  // kv = x @ Wkv
  gemm_bf16_wmma<<<dim3(2048 / 64, M / 128), blk, 0, stream>>>(
      xb, WkvT, kvb, nullptr, 1024, 1024, 1024, 2048, 1.0f, 0);
  // hk = relu(kv[:, :1024] @ Wk1)
  gemm_bf16_wmma<<<dim3(512 / 64, M / 128), blk, 0, stream>>>(
      kvb, Wk1T, hk, nullptr, 1024, 2048, 1024, 512, 1.0f, GF_RELU);
  // k = hk @ Wk2
  gemm_bf16_wmma<<<dim3(1024 / 64, M / 128), blk, 0, stream>>>(
      hk, Wk2T, kbuf, nullptr, 512, 512, 512, 1024, 1.0f, 0);
  // hv = relu(kv[:, 1024:] @ Wv1)
  gemm_bf16_wmma<<<dim3(512 / 64, M / 128), blk, 0, stream>>>(
      kvb + 1024, Wv1T, hv, nullptr, 1024, 2048, 1024, 512, 1.0f, GF_RELU);
  // vT = (hv @ Wv2)^T   (transposed store: vT[h*64+d][b*2048+n])
  gemm_bf16_wmma<<<dim3(1024 / 64, M / 128), blk, 0, stream>>>(
      hv, Wv2T, vTb, nullptr, 512, 512, 512, M, 1.0f, GF_TSTORE);
  // attention (flash, per (b,h) x 64-row blocks)
  mla_attention<<<dim3(32, 2048 / 64), blk, 0, stream>>>(qbuf, kbuf, vTb, ab);
  // out = ab @ Wo + bo  (f32 output)
  gemm_bf16_wmma<<<dim3(1024 / 64, M / 128), blk, 0, stream>>>(
      ab, WoT, out, bo, 1024, 1024, 1024, 1024, 1.0f, GF_F32OUT);
}
